// DecoderWithAttention_28329604284772
// MI455X (gfx1250) — matrix operand order repaired and checked by reading, not compiled, not run on hardware
//
#include <hip/hip_runtime.h>
#include <hip/hip_bf16.h>
#include <cstddef>

typedef __attribute__((ext_vector_type(16))) _Float16 v16h;
typedef __attribute__((ext_vector_type(8)))  float    v8f;

#define B_   128
#define N_   196
#define F_   512
#define E_   512
#define H_   512
#define A_   512
#define V_   30000
#define T_   20   // L-1

// ---------------------------------------------------------------------------
// Generic GEMM: C[M x N] = A[M x K] * W[N x K]^T (+ bias1 + bias2)
// A, W are f32 row-major; converted on the fly to f16; WMMA f32 accumulate.
// One wave computes a 16 x (NSUB*16) tile; NSUB is compile-time so the hot
// loop is branch-free (EXEC stays all-ones around every WMMA, accumulators
// stay pinned in registers).
// Fragment packing follows CDNA5 ISA 7.12.2 VGPR layouts (wave32).
// ---------------------------------------------------------------------------
template <int NSUB>
__global__ __launch_bounds__(256)
void gemm_xwt_f16wmma(const float* __restrict__ A, const float* __restrict__ W,
                      const float* __restrict__ bias1, const float* __restrict__ bias2,
                      float* __restrict__ C,
                      int M, int K, int lda, int ldw, int ldc,
                      int nTiles, int nBase)
{
    const int lane = threadIdx.x & 31;
    const int wave = blockIdx.x * (blockDim.x >> 5) + (threadIdx.x >> 5);

    const int mTiles = M >> 4;
    if (wave >= nTiles * mTiles) return;          // wave-uniform exit
    const int m0 = (wave / nTiles) << 4;
    const int n0 = nBase + (wave % nTiles) * (NSUB * 16);

    const int hi = lane >> 4;       // 0/1 half-wave
    const int ml = lane & 15;       // A row within tile / B column within tile

    const v8f vzero = {0.f,0.f,0.f,0.f,0.f,0.f,0.f,0.f};
    v8f acc[NSUB];
    #pragma unroll
    for (int s = 0; s < NSUB; ++s) acc[s] = vzero;

    const float* arow0 = A + (size_t)(m0 + ml) * lda + hi * 8;
    const float* wrow0 = W + (size_t)(n0 + ml) * ldw + hi * 8;

    for (int k0 = 0; k0 < K; k0 += 32) {
        // ---- A fragment: lane (m=ml, hi) holds K pairs at {0,2,4,6,16,18,20,22}+8*hi
        v16h a;
        const float* arow = arow0 + k0;
        #pragma unroll
        for (int v = 0; v < 8; ++v) {
            const int kb = ((v & 4) << 2) + ((v & 3) << 1); // 0,2,4,6,16,18,20,22
            a[2*v]   = (_Float16)arow[kb];
            a[2*v+1] = (_Float16)arow[kb + 1];
        }
        // ---- NSUB B fragments / WMMAs: lane (n=ml, hi) holds K {0..7,16..23}+8*hi
        #pragma unroll
        for (int s = 0; s < NSUB; ++s) {
            const float* wrow = wrow0 + (size_t)(s * 16) * ldw + k0;
            v16h b;
            #pragma unroll
            for (int v = 0; v < 16; ++v) b[v] = (_Float16)wrow[(v & 7) + ((v >> 3) << 4)];
            acc[s] = __builtin_amdgcn_wmma_f32_16x16x32_f16(
                         false, a, false, b, (short)0, acc[s], false, false);
        }
    }

    // ---- store: lane (n=ml, hi), VGPR r -> D[m0 + r + 8*hi][n]
    #pragma unroll
    for (int s = 0; s < NSUB; ++s) {
        const int n = n0 + s * 16 + ml;
        float bs = 0.f;
        if (bias1) bs += bias1[n];
        if (bias2) bs += bias2[n];
        #pragma unroll
        for (int r = 0; r < 8; ++r)
            C[(size_t)(m0 + r + hi * 8) * ldc + n] = acc[s][r] + bs;
    }
}

// ---------------------------------------------------------------------------
// Wg = [W_ih | W_hh]  (2048 x 1536)
// ---------------------------------------------------------------------------
__global__ void concat_w_kernel(const float* __restrict__ W_ih,
                                const float* __restrict__ W_hh,
                                float* __restrict__ Wg)
{
    size_t idx = (size_t)blockIdx.x * blockDim.x + threadIdx.x;
    if (idx >= (size_t)4 * H_ * (E_ + F_ + H_)) return;
    int n = (int)(idx / (E_ + F_ + H_));
    int k = (int)(idx % (E_ + F_ + H_));
    Wg[idx] = (k < E_ + F_) ? W_ih[(size_t)n * (E_ + F_) + k]
                            : W_hh[(size_t)n * H_ + (k - (E_ + F_))];
}

// ---------------------------------------------------------------------------
// Embedding gather (row 0 forced to zero), emb[B][T][E]
// ---------------------------------------------------------------------------
__global__ void emb_gather_kernel(const int* __restrict__ captions,
                                  const float* __restrict__ embed_W,
                                  float* __restrict__ emb)
{
    size_t idx = (size_t)blockIdx.x * blockDim.x + threadIdx.x;
    if (idx >= (size_t)B_ * T_ * E_) return;
    int e  = (int)(idx & (E_ - 1));
    int bt = (int)(idx >> 9);
    int b = bt / T_, t = bt % T_;
    int tok = captions[b * (T_ + 1) + t];           // captions[:, :-1]
    emb[idx] = (tok == 0) ? 0.f : embed_W[(size_t)tok * E_ + e];
}

// ---------------------------------------------------------------------------
// gfeat[b][f] = mean_n feats[b][n][f]
// ---------------------------------------------------------------------------
__global__ void gfeat_kernel(const float* __restrict__ feats, float* __restrict__ gfeat)
{
    size_t idx = (size_t)blockIdx.x * blockDim.x + threadIdx.x;
    if (idx >= (size_t)B_ * F_) return;
    int b = (int)(idx >> 9), f = (int)(idx & (F_ - 1));
    const float* p = feats + (size_t)b * N_ * F_ + f;
    float s = 0.f;
    for (int n = 0; n < N_; ++n) s += p[(size_t)n * F_];
    gfeat[idx] = s * (1.0f / N_);
}

// ---------------------------------------------------------------------------
// Attention: scores -> softmax -> ctx (f32).  One block per batch element.
// Writes ctx into x[b][512 .. 1023].
// ---------------------------------------------------------------------------
__global__ __launch_bounds__(256)
void attention_kernel(const float* __restrict__ feats, const float* __restrict__ fproj,
                      const float* __restrict__ hproj, const float* __restrict__ v_w,
                      const float* __restrict__ v_b, float* __restrict__ x)
{
    __shared__ float sc[N_];
    __shared__ float red[8];
    const int b = blockIdx.x;
    const int tid = threadIdx.x, lane = tid & 31, wv = tid >> 5;

    // scores[n] = v . tanh(fproj[b,n,:] + hproj[b,:]) + v_b   (one wave per row)
    const float* hp = hproj + (size_t)b * A_;
    for (int n = wv; n < N_; n += 8) {
        const float* fp = fproj + ((size_t)b * N_ + n) * A_;
        float s = 0.f;
        for (int k = lane; k < A_; k += 32) s += v_w[k] * tanhf(fp[k] + hp[k]);
        #pragma unroll
        for (int off = 16; off; off >>= 1) s += __shfl_down(s, off, 32);
        if (lane == 0) sc[n] = s + v_b[0];
    }
    __syncthreads();

    // softmax max
    float m = -1e30f;
    for (int n = tid; n < N_; n += 256) m = fmaxf(m, sc[n]);
    #pragma unroll
    for (int off = 16; off; off >>= 1) m = fmaxf(m, __shfl_down(m, off, 32));
    if (lane == 0) red[wv] = m;
    __syncthreads();
    if (tid == 0) { float mm = red[0]; for (int i = 1; i < 8; ++i) mm = fmaxf(mm, red[i]); red[0] = mm; }
    __syncthreads();
    m = red[0];
    __syncthreads();

    // exp + sum
    float s = 0.f;
    for (int n = tid; n < N_; n += 256) { float e = __expf(sc[n] - m); sc[n] = e; s += e; }
    #pragma unroll
    for (int off = 16; off; off >>= 1) s += __shfl_down(s, off, 32);
    if (lane == 0) red[wv] = s;
    __syncthreads();
    if (tid == 0) { float t = 0.f; for (int i = 0; i < 8; ++i) t += red[i]; red[0] = t; }
    __syncthreads();
    const float inv = 1.0f / red[0];

    // ctx[f] = sum_n w[n] * feats[b,n,f]
    for (int f = tid; f < F_; f += 256) {
        const float* fb = feats + (size_t)b * N_ * F_ + f;
        float acc = 0.f;
        for (int n = 0; n < N_; ++n) acc += sc[n] * fb[(size_t)n * F_];
        x[(size_t)b * (E_ + F_ + H_) + E_ + f] = acc * inv;
    }
}

// ---------------------------------------------------------------------------
// x[b][0:512] = emb[b][t], x[b][1024:1536] = h[b]
// ---------------------------------------------------------------------------
__global__ void pack_x_kernel(const float* __restrict__ emb, const float* __restrict__ h,
                              float* __restrict__ x, int t)
{
    size_t idx = (size_t)blockIdx.x * blockDim.x + threadIdx.x;
    if (idx >= (size_t)B_ * (E_ + H_)) return;
    int b = (int)(idx / (E_ + H_));
    int k = (int)(idx % (E_ + H_));
    float* xr = x + (size_t)b * (E_ + F_ + H_);
    if (k < E_) xr[k] = emb[((size_t)b * T_ + t) * E_ + k];
    else        xr[E_ + F_ + (k - E_)] = h[(size_t)b * H_ + (k - E_)];
}

// ---------------------------------------------------------------------------
// LSTM pointwise: gates[B][4H] -> h, c
// ---------------------------------------------------------------------------
__global__ void lstm_kernel(const float* __restrict__ gates,
                            float* __restrict__ h, float* __restrict__ c)
{
    size_t idx = (size_t)blockIdx.x * blockDim.x + threadIdx.x;
    if (idx >= (size_t)B_ * H_) return;
    int b = (int)(idx >> 9), j = (int)(idx & (H_ - 1));
    const float* g = gates + (size_t)b * 4 * H_;
    float gi = 1.0f / (1.0f + __expf(-g[j]));
    float gf = 1.0f / (1.0f + __expf(-g[H_ + j]));
    float gg = tanhf(g[2 * H_ + j]);
    float go = 1.0f / (1.0f + __expf(-g[3 * H_ + j]));
    float cc = gf * c[idx] + gi * gg;
    c[idx] = cc;
    h[idx] = go * tanhf(cc);
}

// ---------------------------------------------------------------------------
extern "C" void kernel_launch(void* const* d_in, const int* in_sizes, int n_in,
                              void* d_out, int out_size, void* d_ws, size_t ws_size,
                              hipStream_t stream)
{
    const float* feats    = (const float*)d_in[0];
    const int*   captions = (const int*)  d_in[1];
    const float* embed_W  = (const float*)d_in[2];
    const float* Wf_w     = (const float*)d_in[3];
    const float* Wf_b     = (const float*)d_in[4];
    const float* Wh_w     = (const float*)d_in[5];
    const float* Wh_b     = (const float*)d_in[6];
    const float* v_w      = (const float*)d_in[7];
    const float* v_b      = (const float*)d_in[8];
    const float* W_ih     = (const float*)d_in[9];
    const float* W_hh     = (const float*)d_in[10];
    const float* b_ih     = (const float*)d_in[11];
    const float* b_hh     = (const float*)d_in[12];
    const float* init_w   = (const float*)d_in[13];
    const float* init_b   = (const float*)d_in[14];
    const float* initc_w  = (const float*)d_in[15];
    const float* initc_b  = (const float*)d_in[16];
    const float* out_w    = (const float*)d_in[17];
    const float* out_b    = (const float*)d_in[18];
    float* out = (float*)d_out;

    float* ws    = (float*)d_ws;
    float* fproj = ws; ws += (size_t)B_ * N_ * A_;            // 12.85M f
    float* Wg    = ws; ws += (size_t)4 * H_ * (E_ + F_ + H_); // 3.15M f
    float* emb   = ws; ws += (size_t)B_ * T_ * E_;            // 1.31M f
    float* gfeat = ws; ws += (size_t)B_ * F_;
    float* h     = ws; ws += (size_t)B_ * H_;
    float* c     = ws; ws += (size_t)B_ * H_;
    float* hproj = ws; ws += (size_t)B_ * A_;
    float* x     = ws; ws += (size_t)B_ * (E_ + F_ + H_);
    float* gates = ws; ws += (size_t)B_ * 4 * H_;

    // Launch helper: main pass covers floor(N/64) 64-wide tiles with NSUB=4;
    // remainder (N%64, always a multiple of 16 here) gets its own launch.
    auto gemm = [&](const float* A, const float* W, const float* b1, const float* b2,
                    float* C, int M, int Nn, int K, int lda, int ldw, int ldc) {
        const int mT    = M >> 4;
        const int nFull = Nn >> 6;                 // full 64-wide tiles
        if (nFull) {
            int waves  = mT * nFull;
            int blocks = (waves + 7) / 8;          // 8 waves per 256-thread block
            gemm_xwt_f16wmma<4><<<blocks, 256, 0, stream>>>(
                A, W, b1, b2, C, M, K, lda, ldw, ldc, nFull, 0);
        }
        const int rem = (Nn - (nFull << 6)) >> 4;  // leftover 16-wide sub-tiles
        if (rem) {
            int blocks = (mT + 7) / 8;
            switch (rem) {
            case 1: gemm_xwt_f16wmma<1><<<blocks, 256, 0, stream>>>(
                        A, W, b1, b2, C, M, K, lda, ldw, ldc, 1, nFull << 6); break;
            case 2: gemm_xwt_f16wmma<2><<<blocks, 256, 0, stream>>>(
                        A, W, b1, b2, C, M, K, lda, ldw, ldc, 1, nFull << 6); break;
            default: gemm_xwt_f16wmma<3><<<blocks, 256, 0, stream>>>(
                        A, W, b1, b2, C, M, K, lda, ldw, ldc, 1, nFull << 6); break;
            }
        }
    };
    auto ew = [&](size_t n) { return (unsigned)((n + 255) / 256); };

    // ---------------- setup ----------------
    concat_w_kernel<<<ew((size_t)4 * H_ * (E_ + F_ + H_)), 256, 0, stream>>>(W_ih, W_hh, Wg);
    emb_gather_kernel<<<ew((size_t)B_ * T_ * E_), 256, 0, stream>>>(captions, embed_W, emb);
    gfeat_kernel<<<ew((size_t)B_ * F_), 256, 0, stream>>>(feats, gfeat);
    // h0, c0  (small GEMMs on the WMMA path)
    gemm(gfeat, init_w,  init_b,  nullptr, h, B_, H_, F_, F_, F_, H_);
    gemm(gfeat, initc_w, initc_b, nullptr, c, B_, H_, F_, F_, F_, H_);
    // fproj = feats @ Wf^T + Wf_b   (25088 x 512 x 512)
    gemm(feats, Wf_w, Wf_b, nullptr, fproj, B_ * N_, A_, F_, F_, F_, A_);

    // ---------------- time steps ----------------
    for (int t = 0; t < T_; ++t) {
        // hproj = h @ Wh^T + Wh_b
        gemm(h, Wh_w, Wh_b, nullptr, hproj, B_, A_, H_, H_, H_, A_);
        // x = [emb_t, ctx, h]
        pack_x_kernel<<<ew((size_t)B_ * (E_ + H_)), 256, 0, stream>>>(emb, h, x, t);
        attention_kernel<<<B_, 256, 0, stream>>>(feats, fproj, hproj, v_w, v_b, x);
        // gates = x @ Wg^T + b_ih + b_hh   (128 x 2048 x 1536)
        gemm(x, Wg, b_ih, b_hh, gates, B_, 4 * H_, E_ + F_ + H_,
             E_ + F_ + H_, E_ + F_ + H_, 4 * H_);
        lstm_kernel<<<ew((size_t)B_ * H_), 256, 0, stream>>>(gates, h, c);
        // logits_t = h @ out_w^T + out_b  -> out[:, t, :]  (128 x 30000 x 512)
        gemm(h, out_w, out_b, nullptr, out + (size_t)t * V_, B_, V_, H_, H_, H_, (int)((size_t)T_ * V_));
    }
}
